// DeformableConvBlock_28759101014334
// MI455X (gfx1250) — matrix operand order repaired and checked by reading, not compiled, not run on hardware
//
#include <hip/hip_runtime.h>
#include <hip/hip_bf16.h>

// ---- problem constants (from reference) ----
#define B_    4
#define CIN   64
#define COUT  64
#define H_    128
#define W_    128
#define HW    16384      // H_*W_
#define K2C   9          // K*K
#define KTOT  576        // CIN*K2
#define NPOS  65536      // B_*HW
#define EPSC  1e-5f
#define SLOPEC 0.1f

typedef __attribute__((ext_vector_type(16))) __bf16 v16bf;
typedef __attribute__((ext_vector_type(8)))  __bf16 v8bf;
typedef __attribute__((ext_vector_type(8)))  float  v8f;

// ---------------------------------------------------------------------------
// k0: zero stats buffer (sum[64], sumsq[64])
// ---------------------------------------------------------------------------
__global__ void k0_zero(float* __restrict__ sums) {
    if (threadIdx.x < 128) sums[threadIdx.x] = 0.0f;
}

// ---------------------------------------------------------------------------
// k1: offset + modulator 3x3 convs (27 output channels).
// grid = (NPOS/256, 27); blockDim = 256. Weight addresses are wave-uniform.
// ---------------------------------------------------------------------------
__global__ __launch_bounds__(256) void k1_offmod(
    const float* __restrict__ x,
    const float* __restrict__ w_off, const float* __restrict__ b_off,
    const float* __restrict__ w_mod, const float* __restrict__ b_mod,
    float* __restrict__ dyA, float* __restrict__ dxA, float* __restrict__ modA)
{
    const int oc = blockIdx.y;                              // 0..26
    const int p  = blockIdx.x * blockDim.x + threadIdx.x;   // 0..NPOS-1
    const int b  = p >> 14;
    const int hw = p & 16383;
    const int ho = hw >> 7, wo = hw & 127;

    const float* wt  = (oc < 18) ? (w_off + (size_t)oc * KTOT)
                                 : (w_mod + (size_t)(oc - 18) * KTOT);
    const float bias = (oc < 18) ? b_off[oc] : b_mod[oc - 18];
    const float* xb  = x + (size_t)b * CIN * HW;

    float acc = bias;
    for (int c = 0; c < CIN; ++c) {
        const float* xc = xb + c * HW;
        const float* wc = wt + c * 9;
        #pragma unroll
        for (int ky = 0; ky < 3; ++ky) {
            const int y = ho + ky - 1;
            if ((unsigned)y >= (unsigned)H_) continue;
            const float* row = xc + y * W_;
            #pragma unroll
            for (int kx = 0; kx < 3; ++kx) {
                const int xx = wo + kx - 1;
                if ((unsigned)xx < (unsigned)W_) acc += row[xx] * wc[ky * 3 + kx];
            }
        }
    }

    if (oc < 18) {
        const int k2i = oc >> 1;
        float* dst = (oc & 1) ? dxA : dyA;        // off[:,:,0]=dy, off[:,:,1]=dx
        dst[(b * K2C + k2i) * HW + hw] = acc;
    } else {
        modA[(b * K2C + (oc - 18)) * HW + hw] = 2.0f / (1.0f + __expf(-acc));
    }
}

// ---------------------------------------------------------------------------
// k2: fused bilinear gather + modulate + WMMA GEMM (M=64, N=NPOS, K=576).
// 64 threads (2 waves) per block; 64 output pixels per block.
// Wave w owns output channels [32w, 32w+32): 2 WMMA M-tiles.
// ---------------------------------------------------------------------------
__global__ __launch_bounds__(64) void k2_gather_gemm(
    const float* __restrict__ x, const float* __restrict__ w_reg,
    const float* __restrict__ b_reg,
    const float* __restrict__ dyA, const float* __restrict__ dxA,
    const float* __restrict__ modA, float* __restrict__ outRaw)
{
    __shared__ __align__(32) __bf16 wLds[COUT][KTOT];   // 72 KB, K = k2*64 + c
    __shared__ __align__(32) __bf16 sampLds[64][CIN];   // 8 KB, [pixel][c]

    const int t    = threadIdx.x;   // 0..63
    const int lane = t & 31;
    const int wv   = t >> 5;        // wave id 0..1

    // Stage w_reg -> bf16 LDS with K = k2*64 + c ordering. Thread t owns row t.
    for (int k2i = 0; k2i < K2C; ++k2i)
        for (int c = 0; c < CIN; ++c)
            wLds[t][k2i * 64 + c] = (__bf16)w_reg[(size_t)t * KTOT + c * K2C + k2i];
    __syncthreads();

    const int tileBase = blockIdx.x * 64;   // first pixel of this block

    v8f acc[4][2];
    #pragma unroll
    for (int s = 0; s < 4; ++s)
        #pragma unroll
        for (int m = 0; m < 2; ++m)
            #pragma unroll
            for (int r = 0; r < 8; ++r) acc[s][m][r] = 0.0f;

    const int nloc = t & 15;    // pixel-in-subtile for gather stage
    const int cgrp = t >> 4;    // channel group 0..3 (16 channels each)

    for (int k2i = 0; k2i < K2C; ++k2i) {
        const int kh = k2i / 3, kw = k2i % 3;
        __syncthreads();   // previous iteration's WMMA reads done before rewrite

        // ---- gather + modulate stage: fill sampLds[64 pixels][64 channels]
        for (int s = 0; s < 4; ++s) {
            const int p  = tileBase + s * 16 + nloc;
            const int b  = p >> 14;
            const int hw = p & 16383;
            const int ho = hw >> 7, wo = hw & 127;
            const int oi = (b * K2C + k2i) * HW + hw;
            const float dy = dyA[oi], dx = dxA[oi], md = modA[oi];
            const float py = (float)(ho - 1 + kh) + dy;
            const float px = (float)(wo - 1 + kw) + dx;
            const float y0f = floorf(py), x0f = floorf(px);
            const float wy1 = py - y0f, wx1 = px - x0f;
            const float wy0 = 1.0f - wy1, wx0 = 1.0f - wx1;
            const int y0 = (int)y0f, x0 = (int)x0f;
            const int y1 = y0 + 1,  x1 = x0 + 1;
            const bool vy0 = (y0 >= 0) && (y0 < H_), vy1 = (y1 >= 0) && (y1 < H_);
            const bool vx0 = (x0 >= 0) && (x0 < W_), vx1 = (x1 >= 0) && (x1 < W_);
            const bool v00 = vy0 && vx0, v01 = vy0 && vx1;
            const bool v10 = vy1 && vx0, v11 = vy1 && vx1;
            const int yc0 = min(max(y0, 0), H_ - 1), yc1 = min(max(y1, 0), H_ - 1);
            const int xc0 = min(max(x0, 0), W_ - 1), xc1 = min(max(x1, 0), W_ - 1);
            const int o00 = yc0 * W_ + xc0, o01 = yc0 * W_ + xc1;
            const int o10 = yc1 * W_ + xc0, o11 = yc1 * W_ + xc1;
            const float f00 = wy0 * wx0 * md, f01 = wy0 * wx1 * md;
            const float f10 = wy1 * wx0 * md, f11 = wy1 * wx1 * md;
            const float* xb = x + (size_t)b * CIN * HW;
            #pragma unroll
            for (int i = 0; i < 16; ++i) {
                const int c = cgrp * 16 + i;
                const float* xc = xb + c * HW;
                const float g00 = v00 ? xc[o00] : 0.0f;
                const float g01 = v01 ? xc[o01] : 0.0f;
                const float g10 = v10 ? xc[o10] : 0.0f;
                const float g11 = v11 ? xc[o11] : 0.0f;
                sampLds[s * 16 + nloc][c] =
                    (__bf16)(f00 * g00 + f01 * g01 + f10 * g10 + f11 * g11);
            }
        }
        __syncthreads();

        // ---- matrix stage ----
        const int mrow   = lane & 15;
        const int b8     = (lane >> 4) * 16;   // A-frag K block by lane half
        const int base16 = (lane >> 4) * 16;   // B-frag K block by lane half

        v16bf afrag[2][2];                     // [K-chunk][M-tile]
        #pragma unroll
        for (int ch = 0; ch < 2; ++ch) {
            const int cb = k2i * 64 + ch * 32;
            #pragma unroll
            for (int mt = 0; mt < 2; ++mt) {
                const int row = wv * 32 + mt * 16 + mrow;
                v8bf lo = *(const v8bf*)&wLds[row][cb + b8];        // K = cb+b8 .. +7
                v8bf hi = *(const v8bf*)&wLds[row][cb + 8 + b8];    // K = cb+8+b8 ..
                v16bf a;
                #pragma unroll
                for (int i = 0; i < 8; ++i) { a[i] = lo[i]; a[8 + i] = hi[i]; }
                afrag[ch][mt] = a;
            }
        }
        #pragma unroll
        for (int s = 0; s < 4; ++s) {
            #pragma unroll
            for (int ch = 0; ch < 2; ++ch) {
                v16bf bfrag = *(const v16bf*)&sampLds[s * 16 + (lane & 15)][ch * 32 + base16];
                #pragma unroll
                for (int mt = 0; mt < 2; ++mt) {
                    acc[s][mt] = __builtin_amdgcn_wmma_f32_16x16x32_bf16(
                        false, afrag[ch][mt], false, bfrag,
                        (short)0, acc[s][mt], false, false);
                }
            }
        }
    }

    // ---- store D + bias into NCHW raw output (in d_out) ----
    for (int s = 0; s < 4; ++s) {
        const int p  = tileBase + s * 16 + (lane & 15);   // N column = lane%16
        const int b  = p >> 14;
        const int hw = p & 16383;
        for (int mt = 0; mt < 2; ++mt) {
            #pragma unroll
            for (int r = 0; r < 8; ++r) {
                const int M = wv * 32 + mt * 16 + (lane >> 4) * 8 + r;
                outRaw[((size_t)b * COUT + M) * HW + hw] = acc[s][mt][r] + b_reg[M];
            }
        }
    }
}

// ---------------------------------------------------------------------------
// k3: per-channel sum/sumsq over (B,H,W). One block per (b, channel) slab.
// ---------------------------------------------------------------------------
__global__ __launch_bounds__(256) void k3_stats(const float* __restrict__ outRaw,
                                                float* __restrict__ sums)
{
    __shared__ float s1[256], s2[256];
    const int blk = blockIdx.x;          // b*COUT + oc
    const int oc  = blk & 63;
    const float* base = outRaw + (size_t)blk * HW;
    float a = 0.0f, q = 0.0f;
    for (int i = threadIdx.x; i < HW; i += 256) {
        const float v = base[i];
        a += v; q += v * v;
    }
    s1[threadIdx.x] = a; s2[threadIdx.x] = q;
    __syncthreads();
    for (int st = 128; st > 0; st >>= 1) {
        if (threadIdx.x < st) {
            s1[threadIdx.x] += s1[threadIdx.x + st];
            s2[threadIdx.x] += s2[threadIdx.x + st];
        }
        __syncthreads();
    }
    if (threadIdx.x == 0) {
        atomicAdd(&sums[oc],      s1[0]);
        atomicAdd(&sums[64 + oc], s2[0]);
    }
}

// ---------------------------------------------------------------------------
// k4: batchnorm + affine + leaky relu, in place on d_out.
// ---------------------------------------------------------------------------
__global__ __launch_bounds__(256) void k4_norm(float* __restrict__ out,
    const float* __restrict__ sums, const float* __restrict__ gamma,
    const float* __restrict__ beta)
{
    const int i  = blockIdx.x * 256 + threadIdx.x;   // 0 .. B*COUT*HW-1
    const int ch = (i >> 14) & 63;
    const float invN = 1.0f / (float)NPOS;
    const float mean = sums[ch] * invN;
    const float var  = sums[64 + ch] * invN - mean * mean;
    const float v = out[i];
    const float y = (v - mean) * rsqrtf(var + EPSC) * gamma[ch] + beta[ch];
    out[i] = (y >= 0.0f) ? y : SLOPEC * y;
}

// ---------------------------------------------------------------------------
extern "C" void kernel_launch(void* const* d_in, const int* in_sizes, int n_in,
                              void* d_out, int out_size, void* d_ws, size_t ws_size,
                              hipStream_t stream)
{
    (void)in_sizes; (void)n_in; (void)out_size; (void)ws_size;
    const float* x     = (const float*)d_in[0];
    const float* w_off = (const float*)d_in[1];
    const float* b_off = (const float*)d_in[2];
    const float* w_mod = (const float*)d_in[3];
    const float* b_mod = (const float*)d_in[4];
    const float* w_reg = (const float*)d_in[5];
    const float* b_reg = (const float*)d_in[6];
    const float* gamma = (const float*)d_in[7];
    const float* beta  = (const float*)d_in[8];
    float* out = (float*)d_out;

    float* dyA  = (float*)d_ws;                       // B*9*HW floats
    float* dxA  = dyA  + (size_t)B_ * K2C * HW;
    float* modA = dxA  + (size_t)B_ * K2C * HW;
    float* sums = modA + (size_t)B_ * K2C * HW;       // 128 floats

    hipLaunchKernelGGL(k0_zero, dim3(1), dim3(128), 0, stream, sums);
    hipLaunchKernelGGL(k1_offmod, dim3(NPOS / 256, 27), dim3(256), 0, stream,
                       x, w_off, b_off, w_mod, b_mod, dyA, dxA, modA);
    hipLaunchKernelGGL(k2_gather_gemm, dim3(NPOS / 64), dim3(64), 0, stream,
                       x, w_reg, b_reg, dyA, dxA, modA, out);
    hipLaunchKernelGGL(k3_stats, dim3(B_ * COUT), dim3(256), 0, stream, out, sums);
    hipLaunchKernelGGL(k4_norm, dim3((B_ * COUT * HW) / 256), dim3(256), 0, stream,
                       out, sums, gamma, beta);
}
